// CNN_LSTM_36618891165822
// MI455X (gfx1250) — matrix operand order repaired and checked by reading, not compiled, not run on hardware
//
// CNN-LSTM pipeline for MI455X (gfx1250, wave32, WMMA).
// Matmuls (LSTM gates, FC) use v_wmma_f32_16x16x32_bf16 (fp32 accum).
// LSTM: persistent per-batch-tile kernels, h in LDS, c in VGPRs, 77-step loop
// inside the kernel -> no global sync on the sequential chain. Fragment loads
// are double-buffered across K-steps to hide L2 latency behind the WMMAs.
#include <hip/hip_runtime.h>

typedef __attribute__((ext_vector_type(16))) __bf16 v16bf;
typedef __attribute__((ext_vector_type(8)))  float  v8f;

// ---------------- workspace layout (bytes) ----------------
static const size_t OFF_CONST = 0;                          // 512 B savgol consts
static const size_t OFF_YSG   = 512;                        // 256*10000*4
static const size_t OFF_A0    = OFF_YSG + 10240000ull;      // 256*64*624*4
static const size_t OFF_A1    = OFF_A0 + 40894464ull;       // 77*256*128*4
static const size_t OFF_WB    = OFF_A1 + 10092544ull;       // bf16 weights
static const size_t OFF_BIAS  = OFF_WB + 2883584ull;        // 2*1024*4
// aliases (a0 dead after conv1; ysg dead after conv0)
static const size_t OFF_HS0   = OFF_A0;                     // 77*256*256*4
static const size_t OFF_HS1   = OFF_A0 + 20185088ull;
static const size_t OFF_Z0    = OFF_YSG;                    // 256*512*4
static const size_t OFF_Z1    = OFF_YSG + 524288ull;
// bf16 weight sub-offsets
static const size_t WB_WIH0 = 0;        // 1024x128
static const size_t WB_WHH0 = 262144;   // 1024x256
static const size_t WB_WIH1 = 786432;   // 1024x256
static const size_t WB_WHH1 = 1310720;  // 1024x256
static const size_t WB_FC0  = 1835008;  // 512x256
static const size_t WB_FC1  = 2097152;  // 512x512
static const size_t WB_OUT  = 2621440;  // 256x512

// ---------------- WMMA fragment helpers (contiguous k order, same for A and B) ----------------
// A (16x32 bf16, MxK): lane%16 = M row; lanes<16 hold K 0..15,
// lanes>=16 K 16..31.
__device__ __forceinline__ v16bf load_afrag(const float* A, int ld, int kbase, int lane) {
  const int r    = lane & 15;
  const int koff = (lane >> 4) << 4;
  const float* p = A + (size_t)r * ld + kbase + koff;
  float4 x0 = *(const float4*)(p + 0);
  float4 x1 = *(const float4*)(p + 4);
  float4 x2 = *(const float4*)(p + 8);
  float4 x3 = *(const float4*)(p + 12);
  v16bf a;
  a[0]=(__bf16)x0.x; a[1]=(__bf16)x0.y; a[2]=(__bf16)x0.z; a[3]=(__bf16)x0.w;
  a[4]=(__bf16)x1.x; a[5]=(__bf16)x1.y; a[6]=(__bf16)x1.z; a[7]=(__bf16)x1.w;
  a[8]=(__bf16)x2.x; a[9]=(__bf16)x2.y; a[10]=(__bf16)x2.z; a[11]=(__bf16)x2.w;
  a[12]=(__bf16)x3.x; a[13]=(__bf16)x3.y; a[14]=(__bf16)x3.z; a[15]=(__bf16)x3.w;
  return a;
}
// B (32x16 bf16, KxN): lane%16 = N col; lanes<16 hold K 0..15, lanes>=16 K 16..31.
// W stored row-major (N x K) bf16 -> 16 contiguous K values = one 32B load.
__device__ __forceinline__ v16bf load_bfrag(const __bf16* W, int ld, int n0, int kbase, int lane) {
  const int n  = n0 + (lane & 15);
  const int kh = kbase + ((lane >> 4) << 4);
  return *(const v16bf*)(W + (size_t)n * ld + kh);
}
__device__ __forceinline__ v8f wmma_bf16(v16bf a, v16bf b, v8f c) {
  return __builtin_amdgcn_wmma_f32_16x16x32_bf16(false, a, false, b, (short)0, c, false, false);
}
__device__ __forceinline__ float sigf(float v) { return 1.f / (1.f + __expf(-v)); }

// Double-buffered accumulation of acc[0..3] += A(16xK) * W[ncol[s]..]^T.
// Next K-step's fragments are loaded before the current WMMAs consume theirs,
// so the s_wait for the loads is covered by 4 in-flight WMMAs.
__device__ __forceinline__ void mm_stream(const float* A, int lda,
                                          const __bf16* W, int ldw,
                                          const int* ncol, int K, int lane,
                                          v8f* acc) {
  v16bf a0 = load_afrag(A, lda, 0, lane);
  v16bf b0[4];
  #pragma unroll
  for (int s = 0; s < 4; ++s) b0[s] = load_bfrag(W, ldw, ncol[s], 0, lane);
  for (int k = 32; k < K; k += 32) {
    v16bf a1 = load_afrag(A, lda, k, lane);
    v16bf b1[4];
    #pragma unroll
    for (int s = 0; s < 4; ++s) b1[s] = load_bfrag(W, ldw, ncol[s], k, lane);
    #pragma unroll
    for (int s = 0; s < 4; ++s) acc[s] = wmma_bf16(a0, b0[s], acc[s]);
    a0 = a1;
    #pragma unroll
    for (int s = 0; s < 4; ++s) b0[s] = b1[s];
  }
  #pragma unroll
  for (int s = 0; s < 4; ++s) acc[s] = wmma_bf16(a0, b0[s], acc[s]);
}

// ---------------- savgol constants (fp64, one thread, once per launch) -------
__device__ void inv4x4(double M[4][4], double Mi[4][4]) {
  double a[4][8];
  for (int i = 0; i < 4; ++i)
    for (int j = 0; j < 4; ++j) { a[i][j] = M[i][j]; a[i][j + 4] = (i == j) ? 1.0 : 0.0; }
  for (int col = 0; col < 4; ++col) {
    int piv = col;
    for (int r = col + 1; r < 4; ++r) if (fabs(a[r][col]) > fabs(a[piv][col])) piv = r;
    if (piv != col)
      for (int j = 0; j < 8; ++j) { double t = a[col][j]; a[col][j] = a[piv][j]; a[piv][j] = t; }
    double d = a[col][col];
    for (int j = 0; j < 8; ++j) a[col][j] /= d;
    for (int r = 0; r < 4; ++r) if (r != col) {
      double f = a[r][col];
      for (int j = 0; j < 8; ++j) a[r][j] -= f * a[col][j];
    }
  }
  for (int i = 0; i < 4; ++i) for (int j = 0; j < 4; ++j) Mi[i][j] = a[i][j + 4];
}

__global__ void k_sg_consts(float* C) {
  if (threadIdx.x != 0 || blockIdx.x != 0) return;
  double V[11][4], M[4][4], Mi[4][4], pe[4][11];
  // edge fit: positions 0..10
  for (int w = 0; w < 11; ++w) { double tt = 1.0; for (int j = 0; j < 4; ++j) { V[w][j] = tt; tt *= (double)w; } }
  for (int i = 0; i < 4; ++i) for (int j = 0; j < 4; ++j) {
    double s = 0; for (int w = 0; w < 11; ++w) s += V[w][i] * V[w][j]; M[i][j] = s;
  }
  inv4x4(M, Mi);
  for (int j = 0; j < 4; ++j) for (int w = 0; w < 11; ++w) {
    double s = 0; for (int i = 0; i < 4; ++i) s += Mi[j][i] * V[w][i]; pe[j][w] = s;
  }
  for (int k = 0; k < 5; ++k) {
    double t0 = (double)k, t1 = (double)(6 + k);
    for (int w = 0; w < 11; ++w) {
      double pf = 0, pl = 0, p0 = 1, p1 = 1;
      for (int j = 0; j < 4; ++j) { pf += pe[j][w] * p0; pl += pe[j][w] * p1; p0 *= t0; p1 *= t1; }
      C[16 + k * 11 + w] = (float)pf;   // P_FIRST[k][w]
      C[71 + k * 11 + w] = (float)pl;   // P_LAST[k][w]
    }
  }
  // interior taps: centered positions -5..5, pinv row 0
  for (int w = 0; w < 11; ++w) { double tt = 1.0, tc = (double)(w - 5); for (int j = 0; j < 4; ++j) { V[w][j] = tt; tt *= tc; } }
  for (int i = 0; i < 4; ++i) for (int j = 0; j < 4; ++j) {
    double s = 0; for (int w = 0; w < 11; ++w) s += V[w][i] * V[w][j]; M[i][j] = s;
  }
  inv4x4(M, Mi);
  for (int w = 0; w < 11; ++w) {
    double s = 0; for (int i = 0; i < 4; ++i) s += Mi[0][i] * V[w][i]; C[w] = (float)s;
  }
}

// ---------------- savgol apply ----------------
__global__ void k_savgol(const float* __restrict__ x, const float* __restrict__ C,
                         float* __restrict__ out) {
  const int idx = blockIdx.x * 256 + threadIdx.x;
  const int b = idx / 10000, p = idx % 10000;
  const float* xx = x + (size_t)b * 10000;
  float r = 0.f;
  if (p >= 5 && p <= 9994) {
    #pragma unroll
    for (int i = 0; i < 11; ++i) r += C[i] * xx[p - 5 + i];
  } else if (p < 5) {
    #pragma unroll
    for (int w = 0; w < 11; ++w) r += C[16 + p * 11 + w] * xx[w];
  } else {
    const int k = p - 9995;
    #pragma unroll
    for (int w = 0; w < 11; ++w) r += C[71 + k * 11 + w] * xx[9989 + w];
  }
  out[idx] = r;
}

// ---------------- conv block 0: conv(1->64,k16,s8)+relu+pool2+BN ----------------
__global__ void k_conv0(const float* __restrict__ ysg, const float* __restrict__ w0,
                        const float* __restrict__ cb, const float* __restrict__ g,
                        const float* __restrict__ be, const float* __restrict__ mn,
                        const float* __restrict__ vr, float* __restrict__ out) {
  const int t = blockIdx.x * 128 + threadIdx.x;
  if (t >= 624) return;
  const int c = blockIdx.y, b = blockIdx.z;
  const float* xx = ysg + (size_t)b * 10000 + 16 * t;
  float w[16];
  #pragma unroll
  for (int i = 0; i < 16; ++i) w[i] = w0[c * 16 + i];
  float acc0 = cb[c], acc1 = cb[c];
  #pragma unroll
  for (int i = 0; i < 16; ++i) { acc0 += w[i] * xx[i]; acc1 += w[i] * xx[i + 8]; }
  float y = fmaxf(fmaxf(acc0, acc1), 0.f);
  const float inv = rsqrtf(vr[c] + 1e-5f);
  out[((size_t)b * 64 + c) * 624 + t] = g[c] * (y - mn[c]) * inv + be[c];
}

// ---------------- conv block 1: conv(64->128,k8,s4)+relu+pool2+BN -> (T,B,C) --
__global__ void __launch_bounds__(128) k_conv1(const float* __restrict__ a0,
    const float* __restrict__ w1, const float* __restrict__ cb,
    const float* __restrict__ g, const float* __restrict__ be,
    const float* __restrict__ mn, const float* __restrict__ vr,
    float* __restrict__ out) {
  __shared__ float act[64][12];
  const int t = blockIdx.x, b = blockIdx.y, c = threadIdx.x;
  for (int e = threadIdx.x; e < 768; e += 128) {
    int ci = e / 12, off = e % 12;
    act[ci][off] = a0[((size_t)b * 64 + ci) * 624 + 8 * t + off];
  }
  __syncthreads();
  float s0 = cb[c], s1 = cb[c];
  const float* wr = w1 + (size_t)c * 512;
  for (int ci = 0; ci < 64; ++ci) {
    #pragma unroll
    for (int i = 0; i < 8; ++i) {
      float w = wr[ci * 8 + i];
      s0 += w * act[ci][i];
      s1 += w * act[ci][i + 4];
    }
  }
  float y = fmaxf(fmaxf(s0, s1), 0.f);
  const float inv = rsqrtf(vr[c] + 1e-5f);
  out[((size_t)t * 256 + b) * 128 + c] = g[c] * (y - mn[c]) * inv + be[c];
}

// ---------------- weight prep ----------------
__global__ void k_f2bf(const float* __restrict__ in, __bf16* __restrict__ out, int n) {
  int i = blockIdx.x * 256 + threadIdx.x;
  if (i < n) out[i] = (__bf16)in[i];
}
__global__ void k_addv(const float* __restrict__ a, const float* __restrict__ b,
                       float* __restrict__ o, int n) {
  int i = blockIdx.x * 256 + threadIdx.x;
  if (i < n) o[i] = a[i] + b[i];
}

// ---------------- persistent LSTM layer ----------------
// grid = 16 blocks (batch tiles of 16), 512 threads = 16 waves.
// Wave w owns gate columns {g*256 + 16w + j : g in i,f,g,o; j in [0,16)}:
// subtile s == gate s, so i/f/g/o for one hidden unit sit in acc[0..3] of the
// same wave -> the nonlinear update runs entirely in registers; c stays in
// VGPRs (8 per thread). h lives in LDS (16 KB); no gate staging needed.
__global__ void __launch_bounds__(512) k_lstm(const float* __restrict__ seqIn, int Cin,
    const __bf16* __restrict__ Wih, const __bf16* __restrict__ Whh,
    const float* __restrict__ bias, float* __restrict__ hsOut, int T) {
  __shared__ float lds_h[16 * 256];
  const int tid = threadIdx.x;
  const int lane = tid & 31, wave = tid >> 5;   // wave 0..15
  const int rr = lane & 15, hi = lane >> 4;
  const int b0 = blockIdx.x * 16;
  for (int i = tid; i < 4096; i += 512) lds_h[i] = 0.f;
  float creg[8];
  #pragma unroll
  for (int q = 0; q < 8; ++q) creg[q] = 0.f;
  int ncol[4];
  #pragma unroll
  for (int s = 0; s < 4; ++s) ncol[s] = s * 256 + wave * 16;
  __syncthreads();

  for (int t = 0; t < T; ++t) {
    __builtin_prefetch((const void*)(Whh + (size_t)ncol[0] * 256), 0, 1);
    v8f acc[4];
    #pragma unroll
    for (int s = 0; s < 4; ++s) {
      float bv = bias[ncol[s] + rr];
      #pragma unroll
      for (int j = 0; j < 8; ++j) acc[s][j] = bv;
    }
    // input projection: x_t (16 x Cin) @ Wih^T
    const float* At = seqIn + ((size_t)t * 256 + b0) * Cin;
    mm_stream(At, Cin, Wih, Cin, ncol, Cin, lane, acc);
    // recurrent: h (16 x 256, LDS) @ Whh^T
    mm_stream(&lds_h[0], 256, Whh, 256, ncol, 256, lane, acc);
    __syncthreads();   // everyone done reading old h
    #pragma unroll
    for (int j = 0; j < 8; ++j) {
      const float gi = acc[0][j];
      const float gf = acc[1][j];
      const float gg = acc[2][j];
      const float go = acc[3][j];
      const float cn = sigf(gf) * creg[j] + sigf(gi) * tanhf(gg);
      const float hn = sigf(go) * tanhf(cn);
      creg[j] = cn;
      const int m = j + 8 * hi;
      const int col = wave * 16 + rr;
      lds_h[m * 256 + col] = hn;
      hsOut[((size_t)t * 256 + b0 + m) * 256 + col] = hn;
    }
    __syncthreads();   // h updated before next step's reads
  }
}

// ---------------- generic FC GEMM: out = act(A f32 @ W_bf16^T + bias) ----------
// grid = (N/64, M/128), 256 threads; each wave -> 16x64 tile, K step 32.
__global__ void __launch_bounds__(256) k_gemm(const float* __restrict__ A,
    const __bf16* __restrict__ W, const float* __restrict__ bias,
    float* __restrict__ out, int N, int K, int relu) {
  const int tid = threadIdx.x;
  const int lane = tid & 31, wave = tid >> 5;
  const int rr = lane & 15, hi = lane >> 4;
  const int m0 = blockIdx.y * 128 + wave * 16;
  const int n0 = blockIdx.x * 64;
  int ncol[4];
  #pragma unroll
  for (int s = 0; s < 4; ++s) ncol[s] = n0 + s * 16;
  v8f acc[4];
  #pragma unroll
  for (int s = 0; s < 4; ++s) {
    float bv = bias[ncol[s] + rr];
    #pragma unroll
    for (int j = 0; j < 8; ++j) acc[s][j] = bv;
  }
  mm_stream(A + (size_t)m0 * K, K, W, K, ncol, K, lane, acc);
  #pragma unroll
  for (int s = 0; s < 4; ++s)
    #pragma unroll
    for (int j = 0; j < 8; ++j) {
      float v = acc[s][j];
      if (relu) v = fmaxf(v, 0.f);
      out[(size_t)(m0 + j + 8 * hi) * N + ncol[s] + rr] = v;
    }
}

// ---------------- host ----------------
extern "C" void kernel_launch(void* const* d_in, const int* in_sizes, int n_in,
                              void* d_out, int out_size, void* d_ws, size_t ws_size,
                              hipStream_t stream) {
  (void)in_sizes; (void)n_in; (void)out_size; (void)ws_size;
  const float* x       = (const float*)d_in[0];
  const float* conv_w0 = (const float*)d_in[1];
  const float* conv_b0 = (const float*)d_in[2];
  const float* bn_g0   = (const float*)d_in[3];
  const float* bn_b0   = (const float*)d_in[4];
  const float* bn_m0   = (const float*)d_in[5];
  const float* bn_v0   = (const float*)d_in[6];
  const float* conv_w1 = (const float*)d_in[7];
  const float* conv_b1 = (const float*)d_in[8];
  const float* bn_g1   = (const float*)d_in[9];
  const float* bn_b1   = (const float*)d_in[10];
  const float* bn_m1   = (const float*)d_in[11];
  const float* bn_v1   = (const float*)d_in[12];
  const float* Wih0    = (const float*)d_in[13];
  const float* Whh0    = (const float*)d_in[14];
  const float* bih0    = (const float*)d_in[15];
  const float* bhh0    = (const float*)d_in[16];
  const float* Wih1    = (const float*)d_in[17];
  const float* Whh1    = (const float*)d_in[18];
  const float* bih1    = (const float*)d_in[19];
  const float* bhh1    = (const float*)d_in[20];
  const float* fc0_w   = (const float*)d_in[21];
  const float* fc0_b   = (const float*)d_in[22];
  const float* fc1_w   = (const float*)d_in[23];
  const float* fc1_b   = (const float*)d_in[24];
  const float* out_w   = (const float*)d_in[25];
  const float* out_b   = (const float*)d_in[26];

  char*  ws   = (char*)d_ws;
  float* C    = (float*)(ws + OFF_CONST);
  float* ysg  = (float*)(ws + OFF_YSG);
  float* a0   = (float*)(ws + OFF_A0);
  float* a1   = (float*)(ws + OFF_A1);
  float* hs0  = (float*)(ws + OFF_HS0);
  float* hs1  = (float*)(ws + OFF_HS1);
  float* z0   = (float*)(ws + OFF_Z0);
  float* z1   = (float*)(ws + OFF_Z1);
  __bf16* wih0b = (__bf16*)(ws + OFF_WB + WB_WIH0);
  __bf16* whh0b = (__bf16*)(ws + OFF_WB + WB_WHH0);
  __bf16* wih1b = (__bf16*)(ws + OFF_WB + WB_WIH1);
  __bf16* whh1b = (__bf16*)(ws + OFF_WB + WB_WHH1);
  __bf16* fc0b  = (__bf16*)(ws + OFF_WB + WB_FC0);
  __bf16* fc1b  = (__bf16*)(ws + OFF_WB + WB_FC1);
  __bf16* outwb = (__bf16*)(ws + OFF_WB + WB_OUT);
  float* bias0 = (float*)(ws + OFF_BIAS);
  float* bias1 = (float*)(ws + OFF_BIAS + 4096);

  k_sg_consts<<<1, 1, 0, stream>>>(C);
  k_savgol<<<10000, 256, 0, stream>>>(x, C, ysg);
  k_conv0<<<dim3(5, 64, 256), 128, 0, stream>>>(ysg, conv_w0, conv_b0, bn_g0, bn_b0, bn_m0, bn_v0, a0);
  k_conv1<<<dim3(77, 256), 128, 0, stream>>>(a0, conv_w1, conv_b1, bn_g1, bn_b1, bn_m1, bn_v1, a1);

  k_f2bf<<<512,  256, 0, stream>>>(Wih0, wih0b, 1024 * 128);
  k_f2bf<<<1024, 256, 0, stream>>>(Whh0, whh0b, 1024 * 256);
  k_f2bf<<<1024, 256, 0, stream>>>(Wih1, wih1b, 1024 * 256);
  k_f2bf<<<1024, 256, 0, stream>>>(Whh1, whh1b, 1024 * 256);
  k_f2bf<<<512,  256, 0, stream>>>(fc0_w, fc0b, 512 * 256);
  k_f2bf<<<1024, 256, 0, stream>>>(fc1_w, fc1b, 512 * 512);
  k_f2bf<<<512,  256, 0, stream>>>(out_w, outwb, 256 * 512);
  k_addv<<<4, 256, 0, stream>>>(bih0, bhh0, bias0, 1024);
  k_addv<<<4, 256, 0, stream>>>(bih1, bhh1, bias1, 1024);

  k_lstm<<<16, 512, 0, stream>>>(a1, 128, wih0b, whh0b, bias0, hs0, 77);
  k_lstm<<<16, 512, 0, stream>>>(hs0, 256, wih1b, whh1b, bias1, hs1, 77);

  k_gemm<<<dim3(8, 2), 256, 0, stream>>>(hs1 + (size_t)76 * 256 * 256, fc0b, fc0_b, z0, 512, 256, 1);
  k_gemm<<<dim3(8, 2), 256, 0, stream>>>(z0, fc1b, fc1_b, z1, 512, 512, 1);
  k_gemm<<<dim3(4, 2), 256, 0, stream>>>(z1, outwb, out_b, (float*)d_out, 256, 512, 0);
}
